// PerturbEmbedding_25872882991123
// MI455X (gfx1250) — matrix operand order repaired and checked by reading, not compiled, not run on hardware
//
#include <hip/hip_runtime.h>
#include <math.h>

#define N_NODES 8192
#define BATCH   4
#define EDGES   262144
#define EMBED   64
#define HIDDEN  512
#define OUT_DIM 64
#define LN_EPS  1e-5f

typedef _Float16 v16h __attribute__((ext_vector_type(16)));
typedef float    v8f  __attribute__((ext_vector_type(8)));

// ---------------- prep kernels ----------------

__global__ void pe_zero_u32(unsigned int* __restrict__ p, int n) {
    int i = blockIdx.x * blockDim.x + threadIdx.x;
    if (i < n) p[i] = 0u;
}

// deg[g*N + src] += 1 ; cnt[(g*N+src)*4 + b] += (dst == perturb[b])
__global__ void pe_edge_count(const int* __restrict__ ei, const int* __restrict__ pert,
                              unsigned int* __restrict__ deg, unsigned int* __restrict__ cnt) {
    int idx = blockIdx.x * blockDim.x + threadIdx.x;
    if (idx >= BATCH * EDGES) return;
    int g = idx / EDGES;
    int e = idx - g * EDGES;
    const int* base = ei + (size_t)g * 2 * EDGES;
    int src = base[e];
    int dst = base[EDGES + e];
    atomicAdd(&deg[g * N_NODES + src], 1u);
    int p0 = pert[0], p1 = pert[1], p2 = pert[2], p3 = pert[3];
    unsigned int* c = &cnt[(size_t)(g * N_NODES + src) * 4];
    if (dst == p0) atomicAdd(&c[0], 1u);
    if (dst == p1) atomicAdd(&c[1], 1u);
    if (dst == p2) atomicAdd(&c[2], 1u);
    if (dst == p3) atomicAdd(&c[3], 1u);
}

// H[g][i][e] = deg * proj_b[e] + sum_b cnt[b] * proj_w[e*B + b]   (stored f16)
__global__ void pe_build_H(const unsigned int* __restrict__ deg, const unsigned int* __restrict__ cnt,
                           const float* __restrict__ proj_w, const float* __restrict__ proj_b,
                           _Float16* __restrict__ H) {
    int idx = blockIdx.x * blockDim.x + threadIdx.x;
    if (idx >= BATCH * N_NODES * EMBED) return;
    int gi = idx / EMBED;
    int e  = idx - gi * EMBED;
    float v = (float)deg[gi] * proj_b[e];
    const unsigned int* c = &cnt[(size_t)gi * 4];
    #pragma unroll
    for (int b = 0; b < BATCH; ++b)
        v += (float)c[b] * proj_w[e * BATCH + b];
    H[idx] = (_Float16)v;
}

// W1eff[h][e] = sum_k w1[h][k*64+e]  (f16)
__global__ void pe_w1eff(const float* __restrict__ w1, _Float16* __restrict__ w1eff) {
    int idx = blockIdx.x * blockDim.x + threadIdx.x;
    if (idx >= HIDDEN * EMBED) return;
    int h = idx / EMBED, e = idx - h * EMBED;
    const float* r = w1 + (size_t)h * (4 * EMBED);
    w1eff[idx] = (_Float16)(r[e] + r[EMBED + e] + r[2 * EMBED + e] + r[3 * EMBED + e]);
}

// w2t[k][n] = w2[n][k]  (f16)
__global__ void pe_w2t(const float* __restrict__ w2, _Float16* __restrict__ w2t) {
    int idx = blockIdx.x * blockDim.x + threadIdx.x;
    if (idx >= HIDDEN * OUT_DIM) return;
    int k = idx / OUT_DIM, n = idx - k * OUT_DIM;
    w2t[idx] = (_Float16)(w2[(size_t)n * HIDDEN + k]);
}

// ---------------- fused MLP kernel (one wave32 per block, 16 rows per wave) ----------------
// h = H @ W1eff^T + b1 ; LayerNorm ; exact GELU ; out = h @ w2^T + b2

__global__ __launch_bounds__(32)
void pe_mlp_wmma(const _Float16* __restrict__ H, const _Float16* __restrict__ w1eff,
                 const float* __restrict__ b1, const float* __restrict__ ln_g,
                 const float* __restrict__ ln_b, const _Float16* __restrict__ w2t,
                 const float* __restrict__ b2, float* __restrict__ out) {
    __shared__ float    hbuf[16 * HIDDEN];     // 32 KB: f32 activations for normalize pass
    __shared__ _Float16 hbuf16[16 * HIDDEN];   // 16 KB: post-GELU f16 for GEMM2
    __shared__ float    smean[16], srstd[16];

    const int lane = threadIdx.x & 31;
    const int m0   = blockIdx.x * 16;          // global row block over B*N rows
    const int g    = m0 / N_NODES;
    const int mloc = m0 - g * N_NODES;

    const int arow = lane & 15;                // A-frag: row M = lane%16
    const int grp  = lane >> 4;                // lane-half selects K sub-range
    const int bcol = lane & 15;                // B/C/D frag: col N = lane%16

    const _Float16* Hg = H + ((size_t)g * N_NODES + mloc) * EMBED;

    // warm GEMM2 weights into cache while GEMM1 runs (global_prefetch_b8)
    __builtin_prefetch(w2t + (size_t)lane * (HIDDEN * OUT_DIM / 32), 0, 1);

    // ---- GEMM1: (16x64) x (64x512) via v_wmma_f32_16x16x32_f16 ----
    // Per-lane partial LayerNorm moments accumulated in registers during epilogue:
    // lane holds rows {8*grp + r}, columns {nt*16 + bcol}.
    float s8[8], ss8[8];
    #pragma unroll
    for (int r = 0; r < 8; ++r) { s8[r] = 0.f; ss8[r] = 0.f; }

    for (int nt = 0; nt < HIDDEN / 16; ++nt) {
        v8f acc = {};
        #pragma unroll
        for (int kt = 0; kt < EMBED / 32; ++kt) {
            v16h a, b;
            #pragma unroll
            for (int e = 0; e < 16; ++e) {
                // A operand, this code's contiguous order: lane = M, K = grp*16 + e (matches B)
                int ka = grp * 16 + e;
                a[e] = Hg[arow * EMBED + kt * 32 + ka];
                // B operand, same contiguous order as A: lane = N, K = grp*16 + e
                int kb = grp * 16 + e;
                b[e] = w1eff[(nt * 16 + bcol) * EMBED + kt * 32 + kb];
            }
            acc = __builtin_amdgcn_wmma_f32_16x16x32_f16(false, a, false, b,
                                                         (short)0, acc, false, false);
        }
        float bb = b1[nt * 16 + bcol];
        #pragma unroll
        for (int r = 0; r < 8; ++r) {
            int m = r + 8 * grp;               // C/D layout: row = r + 8*(lane/16)
            int n = nt * 16 + bcol;
            float x = acc[r] + bb;
            hbuf[m * HIDDEN + n] = x;
            s8[r]  += x;
            ss8[r] += x * x;
        }
    }

    // ---- LayerNorm stats: butterfly-reduce the 16 lanes of each half ----
    #pragma unroll
    for (int msk = 1; msk < 16; msk <<= 1) {
        #pragma unroll
        for (int r = 0; r < 8; ++r) {
            s8[r]  += __shfl_xor(s8[r],  msk);
            ss8[r] += __shfl_xor(ss8[r], msk);
        }
    }
    if ((lane & 15) == 0) {
        #pragma unroll
        for (int r = 0; r < 8; ++r) {
            float mu  = s8[r]  * (1.0f / HIDDEN);
            float var = ss8[r] * (1.0f / HIDDEN) - mu * mu;
            smean[8 * grp + r] = mu;
            srstd[8 * grp + r] = rsqrtf(var + LN_EPS);
        }
    }
    __syncthreads();

    // ---- normalize + exact-erf GELU -> f16 ----
    for (int i = lane; i < 16 * HIDDEN; i += 32) {
        int m = i >> 9, n = i & (HIDDEN - 1);
        float x = (hbuf[i] - smean[m]) * srstd[m] * ln_g[n] + ln_b[n];
        x = 0.5f * x * (1.0f + erff(x * 0.70710678118654752f));
        hbuf16[i] = (_Float16)x;
    }
    __syncthreads();

    // ---- GEMM2: (16x512) x (512x64) via v_wmma_f32_16x16x32_f16 ----
    float* og = out + ((size_t)g * N_NODES + mloc) * OUT_DIM;
    for (int ot = 0; ot < OUT_DIM / 16; ++ot) {
        v8f acc = {};
        #pragma unroll
        for (int kt = 0; kt < HIDDEN / 32; ++kt) {
            v16h a, b;
            #pragma unroll
            for (int e = 0; e < 16; ++e) {
                int ka = grp * 16 + e;
                a[e] = hbuf16[arow * HIDDEN + kt * 32 + ka];
                int kb = grp * 16 + e;
                b[e] = w2t[(kt * 32 + kb) * OUT_DIM + ot * 16 + bcol];
            }
            acc = __builtin_amdgcn_wmma_f32_16x16x32_f16(false, a, false, b,
                                                         (short)0, acc, false, false);
        }
        float bb = b2[ot * 16 + bcol];
        #pragma unroll
        for (int r = 0; r < 8; ++r) {
            int m = r + 8 * grp;
            int n = ot * 16 + bcol;
            og[(size_t)m * OUT_DIM + n] = acc[r] + bb;
        }
    }
}

// ---------------- launcher ----------------

extern "C" void kernel_launch(void* const* d_in, const int* in_sizes, int n_in,
                              void* d_out, int out_size, void* d_ws, size_t ws_size,
                              hipStream_t stream) {
    (void)in_sizes; (void)n_in; (void)out_size; (void)ws_size;

    const int*   edge_index = (const int*)d_in[0];
    const int*   perturb    = (const int*)d_in[1];
    const float* proj_w     = (const float*)d_in[2];
    const float* proj_b     = (const float*)d_in[3];
    const float* w1         = (const float*)d_in[4];
    const float* b1         = (const float*)d_in[5];
    const float* ln_g       = (const float*)d_in[6];
    const float* ln_b       = (const float*)d_in[7];
    const float* w2         = (const float*)d_in[8];
    const float* b2         = (const float*)d_in[9];
    float*       out        = (float*)d_out;

    // workspace layout (bytes)
    char* ws = (char*)d_ws;
    unsigned int* deg   = (unsigned int*)(ws + 0);                 // 4*8192*4    = 128 KB
    unsigned int* cnt   = (unsigned int*)(ws + 131072);            // 4*8192*16   = 512 KB
    _Float16*     Hbuf  = (_Float16*)(ws + 655360);                // 4*8192*64*2 = 4 MB
    _Float16*     w1eff = (_Float16*)(ws + 4849664);               // 512*64*2    = 64 KB
    _Float16*     w2t   = (_Float16*)(ws + 4915200);               // 512*64*2    = 64 KB

    const int nz = BATCH * N_NODES * 5;                            // deg + cnt words
    pe_zero_u32<<<(nz + 255) / 256, 256, 0, stream>>>(deg, nz);

    pe_edge_count<<<(BATCH * EDGES + 255) / 256, 256, 0, stream>>>(edge_index, perturb, deg, cnt);

    pe_build_H<<<(BATCH * N_NODES * EMBED + 255) / 256, 256, 0, stream>>>(deg, cnt, proj_w, proj_b, Hbuf);

    pe_w1eff<<<(HIDDEN * EMBED + 255) / 256, 256, 0, stream>>>(w1, w1eff);
    pe_w2t<<<(HIDDEN * OUT_DIM + 255) / 256, 256, 0, stream>>>(w2, w2t);

    // one wave32 per block, 16 rows each, over B*N rows
    pe_mlp_wmma<<<(BATCH * N_NODES) / 16, 32, 0, stream>>>(Hbuf, w1eff, b1, ln_g, ln_b, w2t, b2, out);
}
